// ResLFE_Block_23021024707172
// MI455X (gfx1250) — matrix operand order repaired and checked by reading, not compiled, not run on hardware
//
#include <hip/hip_runtime.h>
#include <hip/hip_bf16.h>
#include <math.h>

typedef __bf16 bf16_t;
typedef bf16_t v16bf __attribute__((ext_vector_type(16)));
typedef float  v8f   __attribute__((ext_vector_type(8)));

#define RB 4
#define NP 16000
#define CC 128
#define KK 16
#define DD 4
#define HH 512
#define RTOT (RB*NP)   /* 64000 rows */
#define EPSV 1e-5f

#define MT 128          /* block tile M */
#define NT 64           /* block tile N */
#define KC 128          /* K chunk staged in LDS */
#define LDSLD (KC + 8)  /* 136 bf16 row stride -> 272B, rotates banks */

__device__ __forceinline__ float gelu_tanh(float x) {
    float x3 = x * x * x;
    return 0.5f * x * (1.0f + tanhf(0.7978845608028654f * (x + 0.044715f * x3)));
}

// generic pointer (known LDS-origin) -> raw 32-bit LDS byte offset
__device__ __forceinline__ unsigned lds_addr_of(const void* p) {
    return (unsigned)(uintptr_t)(__attribute__((address_space(3))) const void*)p;
}

// CDNA5 async copy: global memory -> LDS, 16B per lane, tracked by ASYNCcnt
__device__ __forceinline__ void async_load_b128(unsigned lds_off, const void* gaddr) {
    asm volatile("global_load_async_to_lds_b128 %0, %1, off"
                 :: "v"(lds_off), "v"(gaddr)
                 : "memory");
}

__device__ __forceinline__ void wait_asynccnt0() {
    asm volatile("s_wait_asynccnt 0x0" ::: "memory");
}

// ---------------- prep kernels ----------------

// dst[n*K + k] = bf16(src[k*N + n])   (weights -> transposed bf16)
__global__ void transpose_cast_kernel(const float* __restrict__ src,
                                      bf16_t* __restrict__ dst, int K, int N) {
    int idx = blockIdx.x * blockDim.x + threadIdx.x;
    if (idx >= K * N) return;
    int k = idx / N, n = idx % N;
    dst[(size_t)n * K + k] = (bf16_t)src[idx];
}

__global__ void bn_prep_kernel(const float* __restrict__ g, const float* __restrict__ be,
                               const float* __restrict__ m, const float* __restrict__ v,
                               float* __restrict__ sc, float* __restrict__ sh) {
    int c = threadIdx.x;
    if (c < CC) {
        float s = g[c] * rsqrtf(v[c] + EPSV);
        sc[c] = s;
        sh[c] = be[c] - m[c] * s;
    }
}

__global__ void cast_x_kernel(const float* __restrict__ x, bf16_t* __restrict__ xb) {
    size_t i = (size_t)blockIdx.x * blockDim.x + threadIdx.x;
    if (i < (size_t)RTOT * CC) xb[i] = (bf16_t)x[i];
}

__global__ void add_pe_kernel(float* __restrict__ xcur, const float* __restrict__ pe,
                              bf16_t* __restrict__ xb) {
    size_t i4 = (size_t)blockIdx.x * blockDim.x + threadIdx.x;
    if (i4 >= (size_t)RTOT * CC / 4) return;
    float4 a = ((const float4*)xcur)[i4];
    float4 p = ((const float4*)pe)[i4];
    a.x += p.x; a.y += p.y; a.z += p.z; a.w += p.w;
    ((float4*)xcur)[i4] = a;
    size_t i = i4 * 4;
    xb[i+0] = (bf16_t)a.x; xb[i+1] = (bf16_t)a.y;
    xb[i+2] = (bf16_t)a.z; xb[i+3] = (bf16_t)a.w;
}

// ---------------- WMMA GEMM ----------------
// C[m,n] = sum_k A[m,k] * Bt[n,k], fp32 accumulate via v_wmma_f32_16x16x32_bf16.
// Tiles staged global->LDS with GLOBAL_LOAD_ASYNC_TO_LDS_B128 (ASYNCcnt).
// mode 0: outB[row*ldc+col] = bf16( gelu(acc + bias[col]) )
// mode 1: o = res[row*CC+col] + sc[col]*acc + sh[col]; outF=o (f32), outB=bf16(o)
// mode 2: outF[row*ldc+col] = acc (f32)
__global__ __launch_bounds__(256) void gemm_wmma_kernel(
    const bf16_t* __restrict__ A, int lda,
    const bf16_t* __restrict__ Bt, int ldb,
    int Ktot, int mode, int ldc,
    const float* __restrict__ bias,
    const float* __restrict__ sc, const float* __restrict__ sh,
    const float* __restrict__ res,
    float* __restrict__ outF, bf16_t* __restrict__ outB)
{
    __shared__ bf16_t sA[MT * LDSLD];
    __shared__ bf16_t sB[NT * LDSLD];

    const int t    = threadIdx.x;
    const int lane = t & 31;
    const int hf   = lane >> 4;     // half-wave select
    const int l15  = lane & 15;
    const int wid  = t >> 5;
    const int wm   = wid & 3;       // 4 waves along M
    const int wn   = wid >> 2;      // 2 waves along N
    const size_t m0 = (size_t)blockIdx.x * MT;
    const int n0    = blockIdx.y * NT;

    v8f acc[2][2];
    for (int i = 0; i < 2; i++)
        for (int j = 0; j < 2; j++)
            for (int v = 0; v < 8; v++) acc[i][j][v] = 0.0f;

    for (int kc = 0; kc < Ktot; kc += KC) {
        // stage A tile: 128 rows x 128 K, async 16B copies direct to LDS
        {
            int rip   = t >> 4;   // 0..15
            int chunk = t & 15;   // 0..15
            for (int p = 0; p < 8; p++) {
                int row = p * 16 + rip;
                async_load_b128(lds_addr_of(&sA[row * LDSLD + chunk * 8]),
                                A + (m0 + row) * (size_t)lda + kc + chunk * 8);
            }
            // stage B tile: 64 n-rows x 128 K
            for (int p = 0; p < 4; p++) {
                int idx = p * 256 + t;
                int row = idx >> 4;
                int ch  = idx & 15;
                async_load_b128(lds_addr_of(&sB[row * LDSLD + ch * 8]),
                                Bt + (size_t)(n0 + row) * ldb + kc + ch * 8);
            }
        }
        wait_asynccnt0();
        __syncthreads();

        for (int s = 0; s < 4; s++) {
            int k0 = s * 32;
            v16bf af[2], bq[2];
            // A fragment: row M = l15, K = hf*8 + {0..7} and 16 + hf*8 + {0..7}
            for (int i = 0; i < 2; i++) {
                int mr = wm * 32 + 16 * i + l15;
                const bf16_t* pa = &sA[mr * LDSLD + k0 + hf * 8];
                uint4* r = (uint4*)&af[i];
                r[0] = *(const uint4*)pa;
                r[1] = *(const uint4*)(pa + 16);
            }
            // B fragment: col N = l15, K = hf*8 + {0..7} and 16 + hf*8 + {0..7}
            for (int j = 0; j < 2; j++) {
                int nr = wn * 32 + 16 * j + l15;
                const bf16_t* pb = &sB[nr * LDSLD + k0 + hf * 8];
                uint4* r = (uint4*)&bq[j];
                r[0] = *(const uint4*)pb;
                r[1] = *(const uint4*)(pb + 16);
            }
            for (int i = 0; i < 2; i++)
                for (int j = 0; j < 2; j++)
                    acc[i][j] = __builtin_amdgcn_wmma_f32_16x16x32_bf16(
                        false, af[i], false, bq[j], (short)0, acc[i][j], false, false);
        }
        __syncthreads();
    }

    // epilogue: C/D layout -> M = hf*8 + v, N = l15
    for (int i = 0; i < 2; i++) {
        for (int j = 0; j < 2; j++) {
            int col = n0 + wn * 32 + 16 * j + l15;
            float bia = 0.0f, s_ = 0.0f, h_ = 0.0f;
            if (mode == 0) bia = bias[col];
            if (mode == 1) { s_ = sc[col]; h_ = sh[col]; }
            for (int v = 0; v < 8; v++) {
                size_t row = m0 + wm * 32 + 16 * i + hf * 8 + v;
                float a = acc[i][j][v];
                if (mode == 0) {
                    outB[row * (size_t)ldc + col] = (bf16_t)gelu_tanh(a + bia);
                } else if (mode == 1) {
                    float o = res[row * (size_t)CC + col] + s_ * a + h_;
                    outF[row * (size_t)CC + col] = o;
                    outB[row * (size_t)CC + col] = (bf16_t)o;
                } else {
                    outF[row * (size_t)ldc + col] = a;
                }
            }
        }
    }
}

// ---------------- KNN gather-max + BN + residual ----------------
// one wave per row: 32 lanes x float4 covers 128 channels; y is L2-resident
__global__ __launch_bounds__(256) void vfr_gather_kernel(
    const float* __restrict__ y, const int* __restrict__ knn,
    const float* __restrict__ sc, const float* __restrict__ sh,
    float* __restrict__ xcur, bf16_t* __restrict__ xb)
{
    int wid = threadIdx.x >> 5, lane = threadIdx.x & 31;
    size_t r = (size_t)blockIdx.x * 8 + wid;
    if (r >= (size_t)RTOT) return;
    int b = (int)(r / NP), n = (int)(r % NP);
    float4 own = ((const float4*)(y + r * CC))[lane];
    float4 mx = make_float4(-3.4e38f, -3.4e38f, -3.4e38f, -3.4e38f);
    const int* kn = knn + ((size_t)b * NP + n) * KK;
#pragma unroll
    for (int k = 0; k < KK; k++) {
        size_t rr = (size_t)b * NP + kn[k];
        if (k + 1 < KK)
            __builtin_prefetch(y + ((size_t)b * NP + kn[k + 1]) * CC + lane * 4, 0, 0);
        float4 g = ((const float4*)(y + rr * CC))[lane];
        mx.x = fmaxf(mx.x, g.x - own.x);
        mx.y = fmaxf(mx.y, g.y - own.y);
        mx.z = fmaxf(mx.z, g.z - own.z);
        mx.w = fmaxf(mx.w, g.w - own.w);
    }
    float4 s4 = ((const float4*)sc)[lane];
    float4 h4 = ((const float4*)sh)[lane];
    float4 xc = ((const float4*)(xcur + r * CC))[lane];
    float4 o;
    o.x = xc.x + s4.x * mx.x + h4.x;
    o.y = xc.y + s4.y * mx.y + h4.y;
    o.z = xc.z + s4.z * mx.z + h4.z;
    o.w = xc.w + s4.w * mx.w + h4.w;
    ((float4*)(xcur + r * CC))[lane] = o;
    bf16_t* xbr = xb + r * CC + lane * 4;
    xbr[0] = (bf16_t)o.x; xbr[1] = (bf16_t)o.y;
    xbr[2] = (bf16_t)o.z; xbr[3] = (bf16_t)o.w;
}

// ---------------- host launch ----------------
extern "C" void kernel_launch(void* const* d_in, const int* in_sizes, int n_in,
                              void* d_out, int out_size, void* d_ws, size_t ws_size,
                              hipStream_t stream) {
    (void)in_sizes; (void)n_in; (void)out_size; (void)ws_size;

    const float* x_in    = (const float*)d_in[0];
    const float* pe      = (const float*)d_in[1];
    const int*   knn     = (const int*)d_in[2];
    const float* mlp_W1  = (const float*)d_in[3];
    const float* mlp_b1  = (const float*)d_in[4];
    const float* mlp_W2  = (const float*)d_in[5];
    const float* mlp_g   = (const float*)d_in[6];
    const float* mlp_be  = (const float*)d_in[7];
    const float* mlp_m   = (const float*)d_in[8];
    const float* mlp_v   = (const float*)d_in[9];
    const float* vfr_W   = (const float*)d_in[10];
    const float* vfr_g   = (const float*)d_in[11];
    const float* vfr_be  = (const float*)d_in[12];
    const float* vfr_m   = (const float*)d_in[13];
    const float* vfr_v   = (const float*)d_in[14];
    const float* ffn_W1  = (const float*)d_in[15];
    const float* ffn_b1  = (const float*)d_in[16];
    const float* ffn_W2  = (const float*)d_in[17];
    const float* ffn_g   = (const float*)d_in[18];
    const float* ffn_be  = (const float*)d_in[19];
    const float* ffn_m   = (const float*)d_in[20];
    const float* ffn_v   = (const float*)d_in[21];

    char* wsp = (char*)d_ws;
    auto alloc = [&](size_t bytes) -> char* {
        char* p = wsp;
        wsp += (bytes + 255) & ~(size_t)255;
        return p;
    };
    bf16_t* xb  = (bf16_t*)alloc((size_t)RTOT * CC * 2);
    bf16_t* hbf = (bf16_t*)alloc((size_t)RTOT * HH * 2);
    float*  ybuf = (float*)alloc((size_t)RTOT * CC * 4);
    bf16_t* w1t = (bf16_t*)alloc((size_t)CC * HH * 2);
    bf16_t* w2t = (bf16_t*)alloc((size_t)HH * CC * 2);
    bf16_t* f1t = (bf16_t*)alloc((size_t)DD * CC * HH * 2);
    bf16_t* f2t = (bf16_t*)alloc((size_t)DD * HH * CC * 2);
    bf16_t* vt  = (bf16_t*)alloc((size_t)DD * CC * CC * 2);
    float* mlp_sc = (float*)alloc(CC * 4);
    float* mlp_sh = (float*)alloc(CC * 4);
    float* ffn_sc = (float*)alloc(DD * CC * 4);
    float* ffn_sh = (float*)alloc(DD * CC * 4);
    float* vfr_sc = (float*)alloc(DD * CC * 4);
    float* vfr_sh = (float*)alloc(DD * CC * 4);
    float* xcur = (float*)d_out;

    dim3 blk(256);

    // weight repack (transpose + bf16 cast)
    transpose_cast_kernel<<<(CC * HH + 255) / 256, blk, 0, stream>>>(mlp_W1, w1t, CC, HH);
    transpose_cast_kernel<<<(HH * CC + 255) / 256, blk, 0, stream>>>(mlp_W2, w2t, HH, CC);
    for (int d = 0; d < DD; d++) {
        transpose_cast_kernel<<<(CC * HH + 255) / 256, blk, 0, stream>>>(
            ffn_W1 + (size_t)d * CC * HH, f1t + (size_t)d * CC * HH, CC, HH);
        transpose_cast_kernel<<<(HH * CC + 255) / 256, blk, 0, stream>>>(
            ffn_W2 + (size_t)d * HH * CC, f2t + (size_t)d * HH * CC, HH, CC);
        transpose_cast_kernel<<<(CC * CC + 255) / 256, blk, 0, stream>>>(
            vfr_W + (size_t)d * CC * CC, vt + (size_t)d * CC * CC, CC, CC);
    }
    // fold BN into scale/shift
    bn_prep_kernel<<<1, 128, 0, stream>>>(mlp_g, mlp_be, mlp_m, mlp_v, mlp_sc, mlp_sh);
    for (int d = 0; d < DD; d++) {
        bn_prep_kernel<<<1, 128, 0, stream>>>(ffn_g + d * CC, ffn_be + d * CC,
                                              ffn_m + d * CC, ffn_v + d * CC,
                                              ffn_sc + d * CC, ffn_sh + d * CC);
        bn_prep_kernel<<<1, 128, 0, stream>>>(vfr_g + d * CC, vfr_be + d * CC,
                                              vfr_m + d * CC, vfr_v + d * CC,
                                              vfr_sc + d * CC, vfr_sh + d * CC);
    }

    cast_x_kernel<<<((size_t)RTOT * CC + 255) / 256, blk, 0, stream>>>(x_in, xb);

    // leading FFN: x = x + BN(gelu(x@W1+b1)@W2)
    gemm_wmma_kernel<<<dim3(RTOT / MT, HH / NT), blk, 0, stream>>>(
        xb, CC, w1t, CC, CC, 0, HH, mlp_b1, nullptr, nullptr, nullptr, nullptr, hbf);
    gemm_wmma_kernel<<<dim3(RTOT / MT, CC / NT), blk, 0, stream>>>(
        hbf, HH, w2t, HH, HH, 1, CC, nullptr, mlp_sc, mlp_sh, x_in, xcur, xb);

    for (int d = 0; d < DD; d++) {
        add_pe_kernel<<<((size_t)RTOT * CC / 4 + 255) / 256, blk, 0, stream>>>(xcur, pe, xb);
        // VFR projection y = a @ W
        gemm_wmma_kernel<<<dim3(RTOT / MT, CC / NT), blk, 0, stream>>>(
            xb, CC, vt + (size_t)d * CC * CC, CC, CC, 2, CC,
            nullptr, nullptr, nullptr, nullptr, ybuf, nullptr);
        // edge max-pool + BN + residual
        vfr_gather_kernel<<<RTOT / 8, blk, 0, stream>>>(
            ybuf, knn, vfr_sc + d * CC, vfr_sh + d * CC, xcur, xb);
        // FFN_d
        gemm_wmma_kernel<<<dim3(RTOT / MT, HH / NT), blk, 0, stream>>>(
            xb, CC, f1t + (size_t)d * CC * HH, CC, CC, 0, HH,
            ffn_b1 + (size_t)d * HH, nullptr, nullptr, nullptr, nullptr, hbf);
        gemm_wmma_kernel<<<dim3(RTOT / MT, CC / NT), blk, 0, stream>>>(
            hbf, HH, f2t + (size_t)d * HH * CC, HH, HH, 1, CC,
            nullptr, ffn_sc + d * CC, ffn_sh + d * CC, xcur, xcur, xb);
    }
}
